// PromptDrivenDensityNetwork_2688649527352
// MI455X (gfx1250) — matrix operand order repaired and checked by reading, not compiled, not run on hardware
//
#include <hip/hip_runtime.h>
#include <hip/hip_bf16.h>
#include <math.h>

typedef __attribute__((ext_vector_type(16))) _Float16 v16h;
typedef __attribute__((ext_vector_type(8)))  float    v8f;

#define B_IMG 2
#define NBOX  3
#define CEMB  256
#define HW0   64

// ---------------------------------------------------------------------------
// Density stage (tiny FLOPs): build 6x6 resized filters, correlate, reduce.
// ---------------------------------------------------------------------------
__global__ void build_filt6_kernel(const float* __restrict__ emb,
                                   const float* __restrict__ boxes,
                                   float* __restrict__ filt6) {
  int t = blockIdx.x * blockDim.x + threadIdx.x;
  const int total = B_IMG * NBOX * CEMB * 36;
  if (t >= total) return;
  int j = t % 36;
  int c = (t / 36) % CEMB;
  int n = (t / (36 * CEMB)) % NBOX;
  int b = t / (36 * CEMB * NBOX);
  int jy = j / 6, jx = j % 6;
  int x1 = (int)floorf(boxes[(b * NBOX + n) * 4 + 0] * (1.0f / 16.0f));
  int y1 = (int)floorf(boxes[(b * NBOX + n) * 4 + 1] * (1.0f / 16.0f));
  // bilinear 5 -> 6 resize, half-pixel centers (jax.image.resize 'bilinear')
  float sy = (jy + 0.5f) * (5.0f / 6.0f) - 0.5f;
  float sx = (jx + 0.5f) * (5.0f / 6.0f) - 0.5f;
  int y0 = (int)floorf(sy); float ty = sy - (float)y0;
  int x0 = (int)floorf(sx); float tx = sx - (float)x0;
  int y0c = min(max(y0, 0), 4), y1c = min(max(y0 + 1, 0), 4);
  int x0c = min(max(x0, 0), 4), x1c = min(max(x0 + 1, 0), 4);
  const float* base = emb + (size_t)(b * CEMB + c) * HW0 * HW0;
  float v00 = base[(y1 + y0c) * HW0 + (x1 + x0c)];
  float v01 = base[(y1 + y0c) * HW0 + (x1 + x1c)];
  float v10 = base[(y1 + y1c) * HW0 + (x1 + x0c)];
  float v11 = base[(y1 + y1c) * HW0 + (x1 + x1c)];
  filt6[t] = (v00 * (1.f - tx) + v01 * tx) * (1.f - ty) +
             (v10 * (1.f - tx) + v11 * tx) * ty;
}

// comb[b][box][p] = 2*map5 + map6 (scales .9 and 1.0 are identical 5x5 maps)
__global__ void corr_kernel(const float* __restrict__ emb,
                            const float* __restrict__ boxes,
                            const float* __restrict__ filt6,
                            float* __restrict__ comb) {
  int t = blockIdx.x * blockDim.x + threadIdx.x;
  const int total = B_IMG * NBOX * HW0 * HW0;
  if (t >= total) return;
  int p = t % (HW0 * HW0);
  int n = (t / (HW0 * HW0)) % NBOX;
  int b = t / (HW0 * HW0 * NBOX);
  int oy = p / HW0, ox = p % HW0;
  int x1 = (int)floorf(boxes[(b * NBOX + n) * 4 + 0] * (1.f / 16.f));
  int y1 = (int)floorf(boxes[(b * NBOX + n) * 4 + 1] * (1.f / 16.f));
  const float* ebase = emb + (size_t)b * CEMB * HW0 * HW0;
  const float* f6 = filt6 + (size_t)((b * NBOX + n) * CEMB) * 36;
  float m5 = 0.f, m6 = 0.f;
  for (int c = 0; c < CEMB; ++c) {
    const float* ec = ebase + (size_t)c * HW0 * HW0;
    #pragma unroll
    for (int ky = 0; ky < 5; ++ky) {
      int iy = oy - 2 + ky;
      if (iy < 0 || iy >= HW0) continue;
      const float* fr = ec + (y1 + ky) * HW0 + x1;
      #pragma unroll
      for (int kx = 0; kx < 5; ++kx) {
        int ix = ox - 2 + kx;
        if (ix < 0 || ix >= HW0) continue;
        m5 += ec[iy * HW0 + ix] * fr[kx];
      }
    }
    const float* fc = f6 + c * 36;
    #pragma unroll
    for (int ky = 0; ky < 6; ++ky) {
      int iy = oy - 3 + ky;
      if (iy < 0 || iy >= HW0) continue;
      #pragma unroll
      for (int kx = 0; kx < 6; ++kx) {
        int ix = ox - 3 + kx;
        if (ix < 0 || ix >= HW0) continue;
        m6 += ec[iy * HW0 + ix] * fc[ky * 6 + kx];
      }
    }
  }
  comb[t] = 2.0f * m5 + m6;
}

__global__ void dm_kernel(const float* __restrict__ comb, float* __restrict__ dm) {
  int t = blockIdx.x * blockDim.x + threadIdx.x;
  if (t >= B_IMG * HW0 * HW0) return;
  int p = t % (HW0 * HW0);
  int b = t / (HW0 * HW0);
  const float* cb = comb + (size_t)b * NBOX * HW0 * HW0;
  dm[t] = (cb[p] + cb[HW0 * HW0 + p] + cb[2 * HW0 * HW0 + p]) * (1.0f / 9.0f);
}

__global__ void att_kernel(const float* __restrict__ emb,
                           const float* __restrict__ dm,
                           float* __restrict__ att) {
  int t = blockIdx.x * blockDim.x + threadIdx.x;
  if (t >= B_IMG * CEMB * HW0 * HW0) return;
  int p = t % (HW0 * HW0);
  int b = t / (CEMB * HW0 * HW0);
  att[t] = emb[t] * dm[b * HW0 * HW0 + p];
}

// ---------------------------------------------------------------------------
// Weight repack: fp32 [O][I][ky][kx]  ->  f16 K-major [O][r*Cin + ci],
// r = ky*k+kx. Contiguous 32-half rows per K-chunk so the WMMA A fragment
// loads straight from global as packed dwords.
// ---------------------------------------------------------------------------
__global__ void pack_w_kernel(const float* __restrict__ w,
                              _Float16* __restrict__ wp,
                              int Cout, int Cin, int k2) {
  int t = blockIdx.x * blockDim.x + threadIdx.x;
  int Ktot = Cin * k2;
  if (t >= Cout * Ktot) return;
  int oc = t / Ktot;
  int rem = t - oc * Ktot;
  int r  = rem / Cin;
  int ci = rem - r * Cin;
  wp[t] = (_Float16)w[(size_t)oc * Ktot + ci * k2 + r];
}

// ---------------------------------------------------------------------------
// Implicit-GEMM conv, WMMA f16 (acc f32).
//   M-tile = MW*16 out channels, N-tile = 256 pixels, K-chunk 32.
//   8 waves = MW(M) x (8/MW)(N); each wave: 1 A fragment (direct from packed
//   f16 global weights), NACC B fragments from LDS, NACC WMMAs per chunk.
//   K order: r (=ky*k+kx) outer, ci inner -> no div/mod in the hot gather.
//   Spatial sizes are powers of two: lw = log2(W), H == W.
//   UPS: fused align-corners x2 bilinear upsample on the input gather.
// ---------------------------------------------------------------------------
template <int MW, bool UPS>
__launch_bounds__(256)
__global__ void conv_wmma_kernel(const float* __restrict__ in,
                                 const unsigned* __restrict__ wp,  // packed f16 pairs
                                 const float* __restrict__ bias,
                                 const float* __restrict__ resid,
                                 float* __restrict__ out,
                                 int Cin, int Cout, int k, int pad,
                                 int lw, int dorelu) {
  constexpr int NW    = 8 / MW;       // waves along N
  constexpr int NSPAN = 256 / NW;     // pixels per wave
  constexpr int NACC  = NSPAN / 16;   // accumulators (WMMA tiles) per wave

  __shared__ _Float16 sB[256][36];    // [pixel][K], stride 36 halves
  const int tid  = threadIdx.x;
  const int lane = tid & 31, wave = tid >> 5;
  const int wm = wave % MW;           // M subtile (16 rows)
  const int wn = wave / MW;           // N span
  const int lm = lane & 15, lg = lane >> 4;
  const int bb = blockIdx.z;
  const int ocbase  = blockIdx.y * (MW * 16);
  const int pixbase = blockIdx.x * 256;
  const int W   = 1 << lw;
  const int HWp = 1 << (2 * lw);
  const int k2   = k * k;
  const int Ktot = Cin * k2;
  const int KtotU = Ktot >> 1;        // dwords per packed weight row

  // fixed per-thread B-gather column (ci offset within chunk) and base row
  const int bcol  = tid & 31;
  const int brow0 = tid >> 5;

  v8f acc[NACC] = {};

  // this lane's packed-f16 weight row for the A fragment
  const unsigned* wrow = wp + (size_t)(ocbase + wm * 16 + lm) * KtotU;

  for (int r = 0; r < k2; ++r) {
    const int ky = r / k, kx = r - ky * k;     // scalar, once per r
    for (int cb = 0; cb < Cin; cb += 32) {
      const int ci = cb + bcol;
      // ---- stage B tile [256 px x 32 K] (gather + f32->f16)
      const float* inb = in + (size_t)(bb * Cin + ci) * (UPS ? (HWp >> 2) : HWp);
      const int lw2 = lw - 1, w2 = W >> 1;
      #pragma unroll 4
      for (int j = 0; j < 32; ++j) {
        int nn = brow0 + (j << 3);
        int p  = pixbase + nn;
        int oy = p >> lw, ox = p & (W - 1);
        int iy = oy - pad + ky, ix = ox - pad + kx;
        float v = 0.0f;
        if (iy >= 0 && iy < W && ix >= 0 && ix < W) {
          if (!UPS) {
            v = inb[(iy << lw) + ix];
          } else {
            // align_corners=True x2 upsample fused into the gather
            float fy = (float)iy * (float)(w2 - 1) / (float)(W - 1);
            float fx = (float)ix * (float)(w2 - 1) / (float)(W - 1);
            int y0 = (int)fy; float ty = fy - (float)y0;
            int x0 = (int)fx; float tx = fx - (float)x0;
            int y1i = min(y0 + 1, w2 - 1);
            int x1i = min(x0 + 1, w2 - 1);
            float v00 = inb[(y0 << lw2) + x0],  v01 = inb[(y0 << lw2) + x1i];
            float v10 = inb[(y1i << lw2) + x0], v11 = inb[(y1i << lw2) + x1i];
            v = (v00 * (1.f - tx) + v01 * tx) * (1.f - ty) +
                (v10 * (1.f - tx) + v11 * tx) * ty;
          }
        }
        sB[nn][bcol] = (_Float16)v;
      }
      __syncthreads();

      // ---- A fragment direct from global packed f16 (ISA 7.12.2 A layout)
      const int kc = (r * Cin + cb) >> 1;        // dword offset of chunk
      union { v16h v; unsigned u[8]; } af;
      #pragma unroll
      for (int j = 0; j < 4; ++j) {
        af.u[j]     = wrow[kc + lg * 4 + j];
        af.u[4 + j] = wrow[kc + 8 + lg * 4 + j];
      }
      if (kc + 16 < KtotU)
        __builtin_prefetch(&wrow[kc + 16], 0, 1);  // next K-chunk of this row

      // ---- NACC B fragments + WMMAs, reusing the A fragment
      #pragma unroll
      for (int t = 0; t < NACC; ++t) {
        const unsigned* rb = (const unsigned*)(&sB[wn * NSPAN + t * 16 + lm][0]);
        union { v16h v; unsigned u[8]; } bf;
        #pragma unroll
        for (int j = 0; j < 4; ++j) { bf.u[j] = rb[lg * 4 + j]; bf.u[4 + j] = rb[8 + lg * 4 + j]; }
        acc[t] = __builtin_amdgcn_wmma_f32_16x16x32_f16(
            false, af.v, false, bf.v, (short)0, acc[t], false, false);
      }
      __syncthreads();
    }
  }

  // ---- epilogue: bias + optional residual + optional ReLU
  const int pb = pixbase + wn * NSPAN + lm;
  #pragma unroll
  for (int t = 0; t < NACC; ++t) {
    #pragma unroll
    for (int v = 0; v < 8; ++v) {
      int oc = ocbase + wm * 16 + lg * 8 + v;    // C/D layout: M = (lane/16)*8 + v
      float val = acc[t][v] + bias[oc];
      size_t oi = (size_t)(bb * Cout + oc) * HWp + pb + t * 16;
      if (resid)  val += resid[oi];
      if (dorelu) val = fmaxf(val, 0.0f);
      out[oi] = val;
    }
  }
}

// conv5: 32 -> 1 channels, 1x1, relu -> d_out [B,256,256]
__global__ void conv1x1_out_kernel(const float* __restrict__ x,
                                   const float* __restrict__ w,
                                   const float* __restrict__ bias,
                                   float* __restrict__ out, int HWp) {
  int t = blockIdx.x * blockDim.x + threadIdx.x;
  if (t >= B_IMG * HWp) return;
  int p = t % HWp, b = t / HWp;
  float s = bias[0];
  #pragma unroll
  for (int c = 0; c < 32; ++c)
    s += x[(size_t)(b * 32 + c) * HWp + p] * w[c];
  out[t] = fmaxf(s, 0.0f);
}

// ---------------------------------------------------------------------------
extern "C" void kernel_launch(void* const* d_in, const int* in_sizes, int n_in,
                              void* d_out, int out_size, void* d_ws, size_t ws_size,
                              hipStream_t stream) {
  (void)n_in; (void)out_size; (void)ws_size;
  const float* emb   = (const float*)d_in[0];
  const float* boxes = (const float*)d_in[1];

  // params pytree may be flattened in sorted-key (jax) or insertion order.
  bool sorted = (in_sizes[2] == 128 * 256 * 5 * 5);  // conv2.w first => sorted
  #define G(i) ((const float*)d_in[(i)])
  const float *w_c2,*b_c2,*w_c3,*b_c3,*w_c4,*b_c4,*w_c5,*b_c5;
  const float *w_r1a,*b_r1a,*w_r1b,*b_r1b, *w_r2a,*b_r2a,*w_r2b,*b_r2b;
  const float *w_r3a,*b_r3a,*w_r3b,*b_r3b, *w_r4a,*b_r4a,*w_r4b,*b_r4b;
  if (sorted) {
    w_c2=G(2);  b_c2=G(3);  w_c3=G(4);  b_c3=G(5);
    w_c4=G(6);  b_c4=G(7);  w_c5=G(8);  b_c5=G(9);
    w_r1a=G(10); b_r1a=G(11); w_r1b=G(12); b_r1b=G(13);
    w_r2a=G(14); b_r2a=G(15); w_r2b=G(16); b_r2b=G(17);
    w_r3a=G(18); b_r3a=G(19); w_r3b=G(20); b_r3b=G(21);
    w_r4a=G(22); b_r4a=G(23); w_r4b=G(24); b_r4b=G(25);
  } else {
    w_r1a=G(2);  b_r1a=G(3);  w_r1b=G(4);  b_r1b=G(5);
    w_c2=G(6);   b_c2=G(7);
    w_r2a=G(8);  b_r2a=G(9);  w_r2b=G(10); b_r2b=G(11);
    w_c3=G(12);  b_c3=G(13);
    w_r3a=G(14); b_r3a=G(15); w_r3b=G(16); b_r3b=G(17);
    w_c4=G(18);  b_c4=G(19);
    w_r4a=G(20); b_r4a=G(21); w_r4b=G(22); b_r4b=G(23);
    w_c5=G(24);  b_c5=G(25);
  }
  #undef G

  float* ws    = (float*)d_ws;
  float* filt6 = ws;            // 55,296 f
  float* comb  = ws + 65536;    // 24,576 f
  float* dm    = ws + 98304;    //  8,192 f
  const size_t BUF = 8388608;   // 8.4M floats each (max act = [2,64,256,256])
  float* buf1 = ws + 131072;
  float* buf2 = buf1 + BUF;
  float* buf3 = buf2 + BUF;
  _Float16* wh = (_Float16*)(buf3 + BUF);   // packed f16 weights region

  // carve packed-weight buffers (element counts all even -> 4B aligned)
  _Float16* p_r1a = wh;                 // 256*256*49 = 3,211,264
  _Float16* p_r1b = p_r1a + 3211264;
  _Float16* p_c2  = p_r1b + 3211264;    // 128*256*25 =   819,200
  _Float16* p_r2a = p_c2  + 819200;     // 128*128*25 =   409,600
  _Float16* p_r2b = p_r2a + 409600;
  _Float16* p_c3  = p_r2b + 409600;     // 64*128*9   =    73,728
  _Float16* p_r3a = p_c3  + 73728;      // 64*64*9    =    36,864
  _Float16* p_r3b = p_r3a + 36864;
  _Float16* p_c4  = p_r3b + 36864;      // 32*64      =     2,048
  _Float16* p_r4a = p_c4  + 2048;       // 32*32      =     1,024
  _Float16* p_r4b = p_r4a + 1024;

  auto pack = [&](const float* w, _Float16* wp, int Cout, int Cin, int k2) {
    int n = Cout * Cin * k2;
    hipLaunchKernelGGL(pack_w_kernel, dim3((n + 255) / 256), dim3(256), 0, stream,
                       w, wp, Cout, Cin, k2);
  };
  pack(w_r1a, p_r1a, 256, 256, 49);
  pack(w_r1b, p_r1b, 256, 256, 49);
  pack(w_c2,  p_c2,  128, 256, 25);
  pack(w_r2a, p_r2a, 128, 128, 25);
  pack(w_r2b, p_r2b, 128, 128, 25);
  pack(w_c3,  p_c3,   64, 128,  9);
  pack(w_r3a, p_r3a,  64,  64,  9);
  pack(w_r3b, p_r3b,  64,  64,  9);
  pack(w_c4,  p_c4,   32,  64,  1);
  pack(w_r4a, p_r4a,  32,  32,  1);
  pack(w_r4b, p_r4b,  32,  32,  1);

  // density + attention
  hipLaunchKernelGGL(build_filt6_kernel, dim3(216),  dim3(256), 0, stream, emb, boxes, filt6);
  hipLaunchKernelGGL(corr_kernel,        dim3(96),   dim3(256), 0, stream, emb, boxes, filt6, comb);
  hipLaunchKernelGGL(dm_kernel,          dim3(32),   dim3(256), 0, stream, comb, dm);
  hipLaunchKernelGGL(att_kernel,         dim3(8192), dim3(256), 0, stream, emb, dm, buf1);

  // conv dispatcher: big layers (Cout>=64) use the 64-row M-tile instance.
  auto conv = [&](const float* inp, const _Float16* wpk, const float* bs,
                  const float* res, float* outp,
                  int Cin, int Cout, int k, int pad, int lw, int ups, int rl) {
    if (Cout >= 64) {
      dim3 grid((unsigned)(1 << (2 * lw - 8)), (unsigned)(Cout / 64), (unsigned)B_IMG);
      if (ups)
        hipLaunchKernelGGL((conv_wmma_kernel<4, true>), grid, dim3(256), 0, stream,
                           inp, (const unsigned*)wpk, bs, res, outp, Cin, Cout, k, pad, lw, rl);
      else
        hipLaunchKernelGGL((conv_wmma_kernel<4, false>), grid, dim3(256), 0, stream,
                           inp, (const unsigned*)wpk, bs, res, outp, Cin, Cout, k, pad, lw, rl);
    } else {
      dim3 grid((unsigned)(1 << (2 * lw - 8)), (unsigned)(Cout / 32), (unsigned)B_IMG);
      hipLaunchKernelGGL((conv_wmma_kernel<2, false>), grid, dim3(256), 0, stream,
                         inp, (const unsigned*)wpk, bs, res, outp, Cin, Cout, k, pad, lw, rl);
    }
  };

  // res1 @64x64, 256ch, 7x7
  conv(buf1, p_r1a, b_r1a, nullptr, buf2, 256, 256, 7, 3, 6, 0, 1);
  conv(buf2, p_r1b, b_r1b, buf1,    buf3, 256, 256, 7, 3, 6, 0, 1);
  // up x2 fused + conv2 @128x128, 256->128, 5x5
  conv(buf3, p_c2,  b_c2,  nullptr, buf1, 256, 128, 5, 2, 7, 1, 1);
  // res2 @128x128, 128ch, 5x5
  conv(buf1, p_r2a, b_r2a, nullptr, buf2, 128, 128, 5, 2, 7, 0, 1);
  conv(buf2, p_r2b, b_r2b, buf1,    buf3, 128, 128, 5, 2, 7, 0, 1);
  // up x2 fused + conv3 @256x256, 128->64, 3x3
  conv(buf3, p_c3,  b_c3,  nullptr, buf1, 128, 64, 3, 1, 8, 1, 1);
  // res3 @256x256, 64ch, 3x3
  conv(buf1, p_r3a, b_r3a, nullptr, buf2, 64, 64, 3, 1, 8, 0, 1);
  conv(buf2, p_r3b, b_r3b, buf1,    buf3, 64, 64, 3, 1, 8, 0, 1);
  // conv4 1x1, 64->32
  conv(buf3, p_c4,  b_c4,  nullptr, buf1, 64, 32, 1, 0, 8, 0, 1);
  // res4 1x1, 32ch
  conv(buf1, p_r4a, b_r4a, nullptr, buf2, 32, 32, 1, 0, 8, 0, 1);
  conv(buf2, p_r4b, b_r4b, buf1,    buf3, 32, 32, 1, 0, 8, 0, 1);
  // conv5 -> d_out
  hipLaunchKernelGGL(conv1x1_out_kernel, dim3((B_IMG * 256 * 256 + 255) / 256), dim3(256),
                     0, stream, buf3, w_c5, b_c5, (float*)d_out, 256 * 256);
}
